// MLA_8658654068954
// MI455X (gfx1250) — matrix operand order repaired and checked by reading, not compiled, not run on hardware
//
#include <hip/hip_runtime.h>
#include <hip/hip_bf16.h>

#define H_    2048
#define NH_   16
#define QL_   1536
#define KVL_  512
#define NOPE_ 128
#define ROPE_ 64
#define VD_   128
#define QKH_  192
#define B_    2
#define S_    2048
#define MROWS (B_*S_)          // 4096
#define EPS_  1e-6f

typedef __bf16 bf16;
typedef __attribute__((ext_vector_type(16))) __bf16 bf16x16;
typedef __attribute__((ext_vector_type(8)))  float  floatx8;
typedef __attribute__((ext_vector_type(4)))  unsigned uint32x4;
typedef __attribute__((ext_vector_type(4)))  int      int32x4;
typedef __attribute__((ext_vector_type(8)))  int      int32x8;

#if __has_builtin(__builtin_amdgcn_tensor_load_to_lds)
#define HAVE_TDM 1
#endif
#if __has_builtin(__builtin_amdgcn_global_load_async_to_lds_b128)
#define HAVE_ASYNC 1
#endif

__device__ __forceinline__ bf16 f2bf(float f) {
    union { float f; unsigned u; } a; a.f = f;
    unsigned r = a.u + 0x7FFFu + ((a.u >> 16) & 1u);
    union { unsigned short s; bf16 b; } o; o.s = (unsigned short)(r >> 16);
    return o.b;
}

// ---- CDNA5 async / tensor data-mover helpers ------------------------------
__device__ __forceinline__ void async_copy_b128(const void* g, void* l) {
#ifdef HAVE_ASYNC
    __builtin_amdgcn_global_load_async_to_lds_b128((int32x4*)g, (int32x4*)l, 0, 0);
#else
    *(uint4*)l = *(const uint4*)g;
#endif
}

__device__ __forceinline__ void async_wait() {
#ifdef HAVE_ASYNC
#if __has_builtin(__builtin_amdgcn_s_wait_asynccnt)
    __builtin_amdgcn_s_wait_asynccnt(0);
#elif defined(__AMDGCN__)
    asm volatile("s_wait_asynccnt 0x0" ::: "memory");
#endif
#endif
}

__device__ __forceinline__ void tensor_wait() {
#ifdef HAVE_TDM
#if __has_builtin(__builtin_amdgcn_s_wait_tensorcnt)
    __builtin_amdgcn_s_wait_tensorcnt(0);
#elif defined(__AMDGCN__)
    asm volatile("s_wait_tensorcnt 0x0" ::: "memory");
#endif
#endif
}

#ifdef HAVE_TDM
// 2D tile load via Tensor Data Mover. Dims/strides in elements (2-byte data),
// tensor sized exactly to the tile (no OOB clipping needed by the callers).
// Optional LDS padding: +4 DWORDs every 16 DWORDs (i.e. +16B per 64B row).
__device__ __forceinline__ void tdm_load_2d(unsigned lds_off, const bf16* gptr,
                                            unsigned tile_w, unsigned tile_h,
                                            unsigned long long stride_elems,
                                            bool pad)
{
    unsigned long long ga = (unsigned long long)gptr;
    uint32x4 g0;
    g0[0] = 1u;                                    // count=1, user descriptor
    g0[1] = lds_off;                               // lds_addr (bytes)
    g0[2] = (unsigned)(ga & 0xFFFFFFFFu);          // global_addr[31:0]
    g0[3] = (unsigned)((ga >> 32) & 0x1FFFFFFu)    // global_addr[56:32]
          | (2u << 30);                            // type = 2 ("image")
    unsigned w0 = (1u << 16);                      // data_size = 1 -> 2 bytes
    if (pad) w0 |= (1u << 20) | (3u << 22) | (3u << 25); // pad_en, every 16DW, +4DW
    int32x8 g1;
    g1[0] = (int)w0;                               // workgroup_mask=0 (no cluster)
    g1[1] = (int)((tile_w & 0xFFFFu) << 16);                       // tensor_dim0 lo
    g1[2] = (int)(((tile_w >> 16) & 0xFFFFu) | ((tile_h & 0xFFFFu) << 16)); // dim0 hi | dim1 lo
    g1[3] = (int)(((tile_h >> 16) & 0xFFFFu) | ((tile_w & 0xFFFFu) << 16)); // dim1 hi | tile_dim0
    g1[4] = (int)(tile_h & 0xFFFFu);               // tile_dim1 ; tile_dim2 = 0
    g1[5] = (int)(unsigned)(stride_elems & 0xFFFFFFFFull);         // dim0_stride lo
    g1[6] = (int)(unsigned)((stride_elems >> 32) & 0xFFFFull);     // dim0_stride hi
    g1[7] = 0;
    int32x4 z4 = {0, 0, 0, 0};
#if __clang_major__ >= 23
    int32x8 z8 = {0, 0, 0, 0, 0, 0, 0, 0};
    __builtin_amdgcn_tensor_load_to_lds(g0, g1, z4, z4, z8, 0);
#else
    __builtin_amdgcn_tensor_load_to_lds(g0, g1, z4, z4, 0);
#endif
}
#endif // HAVE_TDM

// ---- WMMA fragment loaders (CDNA5 16-bit layouts, wave32) -----------------
__device__ __forceinline__ bf16x16 load_frag_a(const bf16* base, int ld) {
    int lane = threadIdx.x & 31;
    const bf16* p = base + (size_t)(lane & 15) * ld + ((lane >> 4) << 4);
    union { bf16x16 v; uint4 q[2]; } u;
    u.q[0] = *(const uint4*)(p);
    u.q[1] = *(const uint4*)(p + 8);
    return u.v;
}
__device__ __forceinline__ bf16x16 load_frag_b(const bf16* base, int ld) {
    int lane = threadIdx.x & 31;
    const bf16* p = base + (size_t)(lane & 15) * ld + ((lane >> 4) << 4);
    union { bf16x16 v; uint4 q[2]; } u;
    u.q[0] = *(const uint4*)(p);
    u.q[1] = *(const uint4*)(p + 8);
    return u.v;
}

// ---- generic batched WMMA GEMM: C[z] = A[z] * W[z]^T (+bias) * scale ------
// A: M x K (lda), W: N x K (ldw), C: M x N (ldc). Tile 128x64x32, 8 waves,
// TDM-fed double-buffered LDS (pad feature generates the +16B row padding).
__global__ __launch_bounds__(256)
void gemm_wmma_kernel(const bf16* __restrict__ A, const bf16* __restrict__ W,
                      const float* __restrict__ bias,
                      float* __restrict__ Cf, bf16* __restrict__ Cb,
                      int M, int N, int K, int lda, int ldw, int ldc,
                      long long sA, long long sW, long long sC, float scale)
{
    __shared__ bf16 As[2][128 * 40];
    __shared__ bf16 Ws[2][64 * 40];

    const int tid  = threadIdx.x;
    const int z    = blockIdx.z;
    const bf16* Az = A + (size_t)z * sA;
    const bf16* Wz = W + (size_t)z * sW;
    const int m0 = blockIdx.y * 128;
    const int n0 = blockIdx.x * 64;
    const int w = tid >> 5, lane = tid & 31;
    const int wm = (w & 3) * 32, wn = (w >> 2) * 32;

    floatx8 acc[2][2] = {};

    const int arow = tid >> 1, achunk = (tid & 1) * 16;
    const int wrow = (tid & 127) >> 1, wchunk = (tid & 1) * 16;
    (void)arow; (void)achunk; (void)wrow; (void)wchunk;

    auto stage = [&](int buf, int k0) {
#ifdef HAVE_TDM
        if (w == 0) {
            tdm_load_2d((unsigned)(unsigned long long)&As[buf][0],
                        Az + (size_t)m0 * lda + k0, 32, 128, (unsigned long long)lda, true);
            tdm_load_2d((unsigned)(unsigned long long)&Ws[buf][0],
                        Wz + (size_t)n0 * ldw + k0, 32, 64, (unsigned long long)ldw, true);
        }
#else
        *(uint4*)(&As[buf][arow * 40 + achunk]) =
            *(const uint4*)(Az + (size_t)(m0 + arow) * lda + k0 + achunk);
        if (tid < 128)
            *(uint4*)(&Ws[buf][wrow * 40 + wchunk]) =
                *(const uint4*)(Wz + (size_t)(n0 + wrow) * ldw + k0 + wchunk);
#endif
    };

    stage(0, 0);
    tensor_wait();
    __syncthreads();

    for (int k0 = 0; k0 < K; k0 += 32) {
        const int cur = (k0 >> 5) & 1;
        if (k0 + 32 < K) stage(cur ^ 1, k0 + 32);   // DMA next tile while computing

        bf16x16 af[2], bfr[2];
        for (int i = 0; i < 2; ++i) af[i]  = load_frag_a(&As[cur][(wm + i * 16) * 40], 40);
        for (int j = 0; j < 2; ++j) bfr[j] = load_frag_b(&Ws[cur][(wn + j * 16) * 40], 40);
        for (int i = 0; i < 2; ++i)
            for (int j = 0; j < 2; ++j)
                acc[i][j] = __builtin_amdgcn_wmma_f32_16x16x32_bf16(
                    false, af[i], false, bfr[j], (short)0, acc[i][j], false, false);

        tensor_wait();
        __syncthreads();
    }

    const int cn = lane & 15, hi = lane >> 4;
    for (int i = 0; i < 2; ++i)
        for (int j = 0; j < 2; ++j) {
            int gcol = n0 + wn + j * 16 + cn;
            float bv = bias ? bias[gcol] : 0.f;
            for (int r = 0; r < 8; ++r) {
                int grow = m0 + wm + i * 16 + r + hi * 8;
                float v = acc[i][j][r] * scale + bv;
                size_t offc = (size_t)z * sC + (size_t)grow * ldc + gcol;
                if (Cb) Cb[offc] = f2bf(v); else Cf[offc] = v;
            }
        }
}

// ---- flash-attention core over the 576-wide concat(q_lat|q_pe) ------------
// block = 64 threads (2 waves), one 16-row q tile per (b,h).
// K-tiles double-buffered via TDM; q tile staged via async global->LDS.
__global__ __launch_bounds__(64)
void mla_attn_kernel(const bf16* __restrict__ qcat, const bf16* __restrict__ kcat,
                     const int* __restrict__ mask, bf16* __restrict__ olat)
{
    extern __shared__ unsigned char smem_raw[];
    bf16*  qs   = (bf16*)smem_raw;            // 16 x 576
    bf16*  ks0  = qs + 16 * 576;              // 2 x (32 x 576)
    bf16*  kvT  = ks0 + 2 * 32 * 576;         // 512 x 32 (transposed V)
    float* Sbuf = (float*)(kvT + 512 * 32);   // 16 x 32
    bf16*  Pbuf = (bf16*)(Sbuf + 16 * 32);    // 16 x 32
    float* mrow = (float*)(Pbuf + 16 * 32);   // 16
    float* lrow = mrow + 16;                  // 16
    float* srow = lrow + 16;                  // 16

    const int tid = threadIdx.x;
    const int lane = tid & 31;
    const int w = tid >> 5;                   // 0..1 : wave id
    const int q0 = blockIdx.x * 16;
    const int h  = blockIdx.y;
    const int b  = blockIdx.z;
    const size_t qrow0 = (size_t)b * S_ + q0;

    const int kend = q0 + 16;
    const int ntiles = (kend + 31) >> 5;

    // stage q tile (16 x 576) via async loads; first key tile via TDM
    for (int idx = tid; idx < 16 * 72; idx += 64) {
        int r = idx / 72, c8 = (idx % 72) * 8;
        async_copy_b128(qcat + (qrow0 + r) * (size_t)(NH_ * 576) + (size_t)h * 576 + c8,
                        qs + r * 576 + c8);
    }
#ifdef HAVE_TDM
    if (w == 0)
        tdm_load_2d((unsigned)(unsigned long long)ks0,
                    kcat + (size_t)(b * S_) * 576, 576, 32, 576ull, false);
#else
    for (int idx = tid; idx < 32 * 72; idx += 64) {
        int t = idx / 72, c8 = (idx % 72) * 8;
        *(uint4*)(ks0 + t * 576 + c8) =
            *(const uint4*)(kcat + (size_t)(b * S_ + t) * 576 + c8);
    }
#endif
    async_wait();
    tensor_wait();
    if (tid < 16) { mrow[tid] = -3.0e38f; lrow[tid] = 0.f; }
    floatx8 acc[16] = {};
    __syncthreads();

    for (int kt = 0; kt < ntiles; ++kt) {
        const int kt0 = kt * 32;
        const int cur = kt & 1;
        bf16* ks = ks0 + cur * (32 * 576);

        // DMA the next key tile into the other buffer while we compute
#ifdef HAVE_TDM
        if (w == 0 && kt + 1 < ntiles)
            tdm_load_2d((unsigned)(unsigned long long)(ks0 + (cur ^ 1) * (32 * 576)),
                        kcat + (size_t)(b * S_ + (kt + 1) * 32) * 576, 576, 32, 576ull, false);
#else
        if (kt + 1 < ntiles)
            for (int idx = tid; idx < 32 * 72; idx += 64) {
                int t = idx / 72, c8 = (idx % 72) * 8;
                *(uint4*)(ks0 + (cur ^ 1) * (32 * 576) + t * 576 + c8) =
                    *(const uint4*)(kcat + (size_t)(b * S_ + (kt + 1) * 32 + t) * 576 + c8);
            }
#endif

        // transposed copy of the latent 512 for PV B-frags
        for (int idx = tid; idx < 32 * 64; idx += 64) {
            int t = idx >> 6, c8 = (idx & 63) * 8;
            const bf16* src = ks + t * 576 + c8;
            for (int e = 0; e < 8; ++e) kvT[(c8 + e) * 32 + t] = src[e];
        }
        __syncthreads();

        // scores: wave w computes keys [w*16, w*16+16), K = 576 -> 18 WMMAs
        floatx8 s = {};
        for (int c0 = 0; c0 < 576; c0 += 32) {
            bf16x16 a  = load_frag_a(qs + c0, 576);
            bf16x16 bb = load_frag_b(ks + (w * 16) * 576 + c0, 576);
            s = __builtin_amdgcn_wmma_f32_16x16x32_bf16(false, a, false, bb, (short)0, s, false, false);
        }
        {
            int cn = lane & 15, hi = lane >> 4;
            for (int r = 0; r < 8; ++r)
                Sbuf[(r + hi * 8) * 32 + w * 16 + cn] = s[r];
        }
        __syncthreads();

        // online softmax, one thread per q row
        if (tid < 16) {
            int row = tid;
            int qg = q0 + row;
            float mold = mrow[row], lold = lrow[row];
            float mx = mold;
            float sv[32];
            for (int t = 0; t < 32; ++t) {
                int kg = kt0 + t;
                float v = Sbuf[row * 32 + t];
                if (kg > qg || kg >= S_ || mask[b * S_ + kg] == 0) v = -3.0e38f;
                sv[t] = v;
                mx = fmaxf(mx, v);
            }
            float sc = __expf(mold - mx);
            float l = lold * sc;
            for (int t = 0; t < 32; ++t) {
                float p = (sv[t] <= -1.0e38f) ? 0.f : __expf(sv[t] - mx);
                Pbuf[row * 32 + t] = f2bf(p);
                l += p;
            }
            mrow[row] = mx; lrow[row] = l; srow[row] = sc;
        }
        __syncthreads();

        // rescale accumulators + PV (each wave owns 256 channels = 16 tiles)
        {
            int hi = lane >> 4;
            float f[8];
            for (int r = 0; r < 8; ++r) f[r] = srow[r + hi * 8];
            for (int nt = 0; nt < 16; ++nt)
                for (int r = 0; r < 8; ++r) acc[nt][r] *= f[r];
            bf16x16 pa = load_frag_a(Pbuf, 32);
            for (int nt = 0; nt < 16; ++nt) {
                bf16x16 bb = load_frag_b(kvT + (w * 256 + nt * 16) * 32, 32);
                acc[nt] = __builtin_amdgcn_wmma_f32_16x16x32_bf16(
                    false, pa, false, bb, (short)0, acc[nt], false, false);
            }
        }
        tensor_wait();
        __syncthreads();
    }

    // finalize: divide by l, write bf16 o_lat
    int cn = lane & 15, hi = lane >> 4;
    float linv[8];
    for (int r = 0; r < 8; ++r) linv[r] = 1.f / lrow[r + hi * 8];
    for (int nt = 0; nt < 16; ++nt) {
        int ch = w * 256 + nt * 16 + cn;
        for (int r = 0; r < 8; ++r) {
            int row = r + hi * 8;
            olat[(qrow0 + row) * (size_t)(NH_ * 512) + (size_t)h * 512 + ch] =
                f2bf(acc[nt][r] * linv[r]);
        }
    }
}

// ---- elementwise kernels --------------------------------------------------
__global__ void cast_kernel(const float* __restrict__ in, bf16* __restrict__ out, long long n) {
    long long i = (long long)blockIdx.x * blockDim.x + threadIdx.x;
    long long st = (long long)gridDim.x * blockDim.x;
    for (; i < n; i += st) out[i] = f2bf(in[i]);
}

__global__ void rmsnorm_kernel(const float* __restrict__ in, const float* __restrict__ wgt,
                               bf16* __restrict__ out, int cols, int ldin, int ldout) {
    __shared__ float red[256];
    int row = blockIdx.x;
    const float* p = in + (size_t)row * ldin;
    float s = 0.f;
    for (int c = threadIdx.x; c < cols; c += 256) { float v = p[c]; s += v * v; }
    red[threadIdx.x] = s; __syncthreads();
    for (int st = 128; st > 0; st >>= 1) {
        if (threadIdx.x < st) red[threadIdx.x] += red[threadIdx.x + st];
        __syncthreads();
    }
    float inv = rsqrtf(red[0] / cols + EPS_);
    for (int c = threadIdx.x; c < cols; c += 256)
        out[(size_t)row * ldout + c] = f2bf(p[c] * inv * wgt[c]);
}

// wkv_bT[h][c][d] = wkv_b[h][d][c]  (d < NOPE)
__global__ void wkvb_transpose_kernel(const float* __restrict__ w, bf16* __restrict__ out) {
    long long n = (long long)NH_ * KVL_ * NOPE_;
    long long i = (long long)blockIdx.x * blockDim.x + threadIdx.x;
    long long st = (long long)gridDim.x * blockDim.x;
    for (; i < n; i += st) {
        int hh = (int)(i / ((long long)KVL_ * NOPE_));
        int rem = (int)(i % ((long long)KVL_ * NOPE_));
        int c = rem / NOPE_, d = rem % NOPE_;
        out[i] = f2bf(w[((size_t)hh * (NOPE_ + VD_) + d) * KVL_ + c]);
    }
}

__device__ __forceinline__ float rope_val(const float* base, int i, int s) {
    float x  = base[i];
    float pr = (i < 32) ? -base[i + 32] : base[i - 32];
    int j = i & 31;
    float ang = (float)s * __powf(10000.f, -(float)(2 * j) / 64.f);
    return x * __cosf(ang) + pr * __sinf(ang);
}

__global__ void rope_q_kernel(const float* __restrict__ q2, bf16* __restrict__ qcat, float scale) {
    long long n = (long long)MROWS * NH_ * ROPE_;
    long long i = (long long)blockIdx.x * blockDim.x + threadIdx.x;
    long long st = (long long)gridDim.x * blockDim.x;
    for (; i < n; i += st) {
        int i64 = (int)(i % ROPE_);
        long long t = i / ROPE_;
        int hh = (int)(t % NH_);
        long long row = t / NH_;
        int s = (int)(row & (S_ - 1));
        const float* base = q2 + row * (NH_ * QKH_) + hh * QKH_ + NOPE_;
        qcat[row * (NH_ * 576) + hh * 576 + 512 + i64] = f2bf(rope_val(base, i64, s) * scale);
    }
}

__global__ void rope_k_kernel(const float* __restrict__ kvf, bf16* __restrict__ kcat) {
    long long n = (long long)MROWS * ROPE_;
    long long i = (long long)blockIdx.x * blockDim.x + threadIdx.x;
    long long st = (long long)gridDim.x * blockDim.x;
    for (; i < n; i += st) {
        int i64 = (int)(i % ROPE_);
        long long row = i / ROPE_;
        int s = (int)(row & (S_ - 1));
        const float* base = kvf + row * 576 + 512;
        kcat[row * 576 + 512 + i64] = f2bf(rope_val(base, i64, s));
    }
}

// ---- host side ------------------------------------------------------------
static inline size_t align256(size_t x) { return (x + 255) & ~(size_t)255; }
static inline int nblocks(long long n) {
    long long b = (n + 255) / 256; if (b > 32768) b = 32768; return (int)b;
}

extern "C" void kernel_launch(void* const* d_in, const int* in_sizes, int n_in,
                              void* d_out, int out_size, void* d_ws, size_t ws_size,
                              hipStream_t stream) {
    (void)in_sizes; (void)n_in; (void)out_size; (void)ws_size;
    const float* x        = (const float*)d_in[0];
    const int*   mask     = (const int*)d_in[1];
    const float* wq_a_w   = (const float*)d_in[2];
    const float* wq_a_b   = (const float*)d_in[3];
    const float* q_norm_w = (const float*)d_in[4];
    const float* wq_b_w   = (const float*)d_in[5];
    const float* wq_b_b   = (const float*)d_in[6];
    const float* wkv_a_w  = (const float*)d_in[7];
    const float* wkv_a_b  = (const float*)d_in[8];
    const float* kv_nrm_w = (const float*)d_in[9];
    const float* wkv_b_w  = (const float*)d_in[10];
    const float* wo_w     = (const float*)d_in[11];
    const float* wo_b     = (const float*)d_in[12];
    float* out = (float*)d_out;

    unsigned char* ws = (unsigned char*)d_ws;
    size_t off = 0;
    auto alloc = [&](size_t bytes) { void* p = ws + off; off = align256(off + bytes); return p; };

    bf16* xb    = (bf16*)alloc((size_t)MROWS * H_ * 2);
    bf16* wqab  = (bf16*)alloc((size_t)QL_ * H_ * 2);
    bf16* wqbb  = (bf16*)alloc((size_t)NH_ * QKH_ * QL_ * 2);
    bf16* wkvab = (bf16*)alloc((size_t)(KVL_ + ROPE_) * H_ * 2);
    bf16* wkvbb = (bf16*)alloc((size_t)NH_ * (NOPE_ + VD_) * KVL_ * 2);
    bf16* wkvbT = (bf16*)alloc((size_t)NH_ * KVL_ * NOPE_ * 2);
    bf16* wob   = (bf16*)alloc((size_t)H_ * NH_ * VD_ * 2);
    float* q1   = (float*)alloc((size_t)MROWS * QL_ * 4);
    bf16* q1n   = (bf16*)alloc((size_t)MROWS * QL_ * 2);
    float* q2   = (float*)alloc((size_t)MROWS * NH_ * QKH_ * 4);
    bf16* q2b   = (bf16*)alloc((size_t)MROWS * NH_ * QKH_ * 2);
    bf16* qcat  = (bf16*)alloc((size_t)MROWS * NH_ * 576 * 2);
    float* kvf  = (float*)alloc((size_t)MROWS * 576 * 4);
    bf16* kcat  = (bf16*)alloc((size_t)MROWS * 576 * 2);
    bf16* olat  = (bf16*)alloc((size_t)MROWS * NH_ * 512 * 2);
    bf16* ohead = (bf16*)alloc((size_t)MROWS * NH_ * VD_ * 2);

    auto cast = [&](const float* in, bf16* o, long long n) {
        cast_kernel<<<nblocks(n), 256, 0, stream>>>(in, o, n);
    };
    auto gemm = [&](const bf16* A, const bf16* W, const float* bias,
                    float* Cf, bf16* Cb, int M, int N, int K,
                    int lda, int ldw, int ldc,
                    long long sA, long long sW, long long sC, int Z, float scale) {
        dim3 g(N / 64, M / 128, Z);
        gemm_wmma_kernel<<<g, 256, 0, stream>>>(A, W, bias, Cf, Cb,
                                                M, N, K, lda, ldw, ldc, sA, sW, sC, scale);
    };

    const float qscale = 0.07216878364870323f; // 1/sqrt(192)

    // precision conversion of activations + weights
    cast(x, xb, (long long)MROWS * H_);
    cast(wq_a_w, wqab, (long long)QL_ * H_);
    cast(wq_b_w, wqbb, (long long)NH_ * QKH_ * QL_);
    cast(wkv_a_w, wkvab, (long long)(KVL_ + ROPE_) * H_);
    cast(wkv_b_w, wkvbb, (long long)NH_ * (NOPE_ + VD_) * KVL_);
    cast(wo_w, wob, (long long)H_ * NH_ * VD_);
    wkvb_transpose_kernel<<<nblocks((long long)NH_ * KVL_ * NOPE_), 256, 0, stream>>>(wkv_b_w, wkvbT);

    // q = rms_norm(x @ wq_a^T + b) @ wq_b^T + b
    gemm(xb, wqab, wq_a_b, q1, nullptr, MROWS, QL_, H_, H_, H_, QL_, 0, 0, 0, 1, 1.f);
    rmsnorm_kernel<<<MROWS, 256, 0, stream>>>(q1, q_norm_w, q1n, QL_, QL_, QL_);
    gemm(q1n, wqbb, wq_b_b, q2, nullptr, MROWS, NH_ * QKH_, QL_, QL_, QL_, NH_ * QKH_, 0, 0, 0, 1, 1.f);
    cast(q2, q2b, (long long)MROWS * NH_ * QKH_);
    rope_q_kernel<<<nblocks((long long)MROWS * NH_ * ROPE_), 256, 0, stream>>>(q2, qcat, qscale);

    // kv path: kv_full = x @ wkv_a^T + b ; rmsnorm(512) ; rope(k_pe)
    gemm(xb, wkvab, wkv_a_b, kvf, nullptr, MROWS, KVL_ + ROPE_, H_, H_, H_, KVL_ + ROPE_, 0, 0, 0, 1, 1.f);
    rmsnorm_kernel<<<MROWS, 256, 0, stream>>>(kvf, kv_nrm_w, kcat, KVL_, 576, 576);
    rope_k_kernel<<<nblocks((long long)MROWS * ROPE_), 256, 0, stream>>>(kvf, kcat);

    // q_lat = q_nope @ wkv_b[:, :NOPE] (per head), scaled by 1/sqrt(192), into qcat[:, :512]
    gemm(q2b, wkvbT, nullptr, nullptr, qcat,
         MROWS, KVL_, NOPE_, NH_ * QKH_, NOPE_, NH_ * 576,
         (long long)QKH_, (long long)KVL_ * NOPE_, 576, NH_, qscale);

    // flash attention over concat(512|64)
    {
        dim3 ga(S_ / 16, NH_, B_);
        size_t smem = (size_t)(16 * 576 + 2 * 32 * 576 + 512 * 32) * 2
                      + (size_t)(16 * 32) * 4 + (size_t)(16 * 32) * 2 + 3 * 16 * 4;
        mla_attn_kernel<<<ga, 64, smem, stream>>>(qcat, kcat, mask, olat);
    }

    // o_head = o_lat @ wkv_b[:, NOPE:]^T (per head)
    gemm(olat, wkvbb + (size_t)NOPE_ * KVL_, nullptr, nullptr, ohead,
         MROWS, VD_, KVL_, NH_ * 512, KVL_, NH_ * VD_,
         512, (long long)(NOPE_ + VD_) * KVL_, VD_, NH_, 1.f);

    // out = o_head @ wo^T + wo_b   (fp32 final)
    gemm(ohead, wob, wo_b, out, nullptr,
         MROWS, H_, NH_ * VD_, NH_ * VD_, NH_ * VD_, H_, 0, 0, 0, 1, 1.f);
}
